// Simp_Model_100_GRU_EncDec_samehidden_Attn_54872502174226
// MI455X (gfx1250) — matrix operand order repaired and checked by reading, not compiled, not run on hardware
//
#include <hip/hip_runtime.h>

typedef _Float16 h16;
typedef __attribute__((ext_vector_type(16))) _Float16 v16h;
typedef __attribute__((ext_vector_type(8)))  _Float16 v8h;
typedef __attribute__((ext_vector_type(8)))  float    v8f;

#define H   100
#define HP  128   // K-pad for H (mult of 32)
#define G   300   // 3*H gate width
#define GP  304   // N-pad for gates (mult of 16)
#define K2  200   // concat input width
#define K2P 224   // K-pad for 200
#define TE  12
#define TD  50

// ---------------- WMMA fragment helpers (wave32, 16x16x32 f16) ----------------

// A fragment: 16x32 f16 tile from LDS buffer [16 rows][KP], row-major.
// ISA layout: lanes 0-15 row M=lane, elems 0..7 -> K=kb..kb+7, elems 8..15 -> K=kb+16..kb+23
//             lanes 16-31 row M=lane-16, K offset +8 / +24.
__device__ __forceinline__ v16h frag_a(const h16* base, int KP, int kb) {
  int lane = threadIdx.x & 31;
  const h16* p = base + (lane & 15) * KP + kb + ((lane >> 4) << 3);
  v8h lo = *(const v8h*)p;        // 16B aligned
  v8h hi = *(const v8h*)(p + 16); // 16B aligned
  return __builtin_shufflevector(lo, hi, 0,1,2,3,4,5,6,7,8,9,10,11,12,13,14,15);
}

// B fragment: 32x16 f16 tile from GLOBAL padded weight [NP rows][KP], row-major W[N][K].
// lanes 0-15 column N = nt*16+lane, K=kb..kb+7 and kb+16..kb+23; lanes 16-31 same column set, K offset +8 / +24.
__device__ __forceinline__ v16h frag_b(const h16* W, int KP, int nt, int kb) {
  int lane = threadIdx.x & 31;
  const h16* p = W + ((size_t)((nt << 4) + (lane & 15))) * KP + kb + ((lane >> 4) << 3);
  v8h lo = *(const v8h*)p;
  v8h hi = *(const v8h*)(p + 16);
  return __builtin_shufflevector(lo, hi, 0,1,2,3,4,5,6,7,8,9,10,11,12,13,14,15);
}

// C = A(16xK, LDS f16) * W^T (W global f16 [NP][KP]) accumulated f32, stored to LDS [16][CP].
// A fragments are hoisted into registers once. N-tiles processed in blocks of 4 with
// distinct B registers so >=8 global_load_b128 are in flight before each WMMA burst
// (software pipelining; only 1 wave/SIMD available to hide L2 latency).
template<int KC>
__device__ __forceinline__ void wave_gemm(const h16* Albs, int KP, const h16* Wg,
                                          float* Clds, int CP, int ntiles) {
  int lane = threadIdx.x & 31;
  int col0 = lane & 15;
  int rb   = (lane >> 4) << 3;
  v16h a[KC];
#pragma unroll
  for (int kc = 0; kc < KC; ++kc) a[kc] = frag_a(Albs, KP, kc * 32);

  int nt = 0;
  for (; nt + 4 <= ntiles; nt += 4) {
    v8f acc0 = {}, acc1 = {}, acc2 = {}, acc3 = {};
#pragma unroll
    for (int kc = 0; kc < KC; ++kc) {
      v16h b0 = frag_b(Wg, KP, nt + 0, kc * 32);
      v16h b1 = frag_b(Wg, KP, nt + 1, kc * 32);
      v16h b2 = frag_b(Wg, KP, nt + 2, kc * 32);
      v16h b3 = frag_b(Wg, KP, nt + 3, kc * 32);
      acc0 = __builtin_amdgcn_wmma_f32_16x16x32_f16(false, a[kc], false, b0, (short)0, acc0, false, false);
      acc1 = __builtin_amdgcn_wmma_f32_16x16x32_f16(false, a[kc], false, b1, (short)0, acc1, false, false);
      acc2 = __builtin_amdgcn_wmma_f32_16x16x32_f16(false, a[kc], false, b2, (short)0, acc2, false, false);
      acc3 = __builtin_amdgcn_wmma_f32_16x16x32_f16(false, a[kc], false, b3, (short)0, acc3, false, false);
    }
#pragma unroll
    for (int v = 0; v < 8; ++v) {
      Clds[(rb + v) * CP + ((nt + 0) << 4) + col0] = acc0[v];
      Clds[(rb + v) * CP + ((nt + 1) << 4) + col0] = acc1[v];
      Clds[(rb + v) * CP + ((nt + 2) << 4) + col0] = acc2[v];
      Clds[(rb + v) * CP + ((nt + 3) << 4) + col0] = acc3[v];
    }
  }
  for (; nt < ntiles; ++nt) {
    v8f acc = {};
#pragma unroll
    for (int kc = 0; kc < KC; ++kc) {
      v16h b = frag_b(Wg, KP, nt, kc * 32);
      acc = __builtin_amdgcn_wmma_f32_16x16x32_f16(false, a[kc], false, b, (short)0, acc, false, false);
    }
#pragma unroll
    for (int v = 0; v < 8; ++v) Clds[(rb + v) * CP + (nt << 4) + col0] = acc[v];
  }
}

__device__ __forceinline__ float sigmf(float x) { return 1.0f / (1.0f + __expf(-x)); }

// GRU elementwise update. gi/gh: LDS [16][GP] f32 (no bias). Updates hf32 [16][H] and
// hf16 [16][HP] (padded f16 A-staging). Optionally writes f16 hidden to global sequence.
__device__ __forceinline__ void gru_elem(const float* gi, const float* gh,
                                         const float* bih, const float* bhh,
                                         float* hf32, h16* hf16,
                                         h16* hseq_out, int ldseq) {
  int lane = threadIdx.x & 31;
  for (int e = lane; e < 16 * H; e += 32) {
    int b = e / H, c = e - b * H;
    float ir = gi[b * GP + c]         + bih[c];
    float iz = gi[b * GP + c + H]     + bih[c + H];
    float in_= gi[b * GP + c + 2 * H] + bih[c + 2 * H];
    float hr = gh[b * GP + c]         + bhh[c];
    float hz = gh[b * GP + c + H]     + bhh[c + H];
    float hn = gh[b * GP + c + 2 * H] + bhh[c + 2 * H];
    float r  = sigmf(ir + hr);
    float z  = sigmf(iz + hz);
    float ng = tanhf(in_ + r * hn);
    float hv = (1.0f - z) * ng + z * hf32[b * H + c];
    hf32[b * H + c]  = hv;
    hf16[b * HP + c] = (h16)hv;
    if (hseq_out) hseq_out[(size_t)b * ldseq + c] = (h16)hv;
  }
}

// ---------------- weight pad+convert to f16 ----------------
__global__ void pad_f16(const float* __restrict__ src, h16* __restrict__ dst,
                        int N, int K, int NP, int KP) {
  int i = blockIdx.x * blockDim.x + threadIdx.x;
  if (i >= NP * KP) return;
  int n = i / KP, k = i - n * KP;
  float v = (n < N && k < K) ? src[n * K + k] : 0.0f;
  dst[i] = (h16)v;
}

// ---------------- one GRU layer over TE steps (one wave per 16-row batch tile) --------
template<typename TI>
__global__ void __launch_bounds__(32)
gru_layer_kernel(const TI* __restrict__ xseq,          // [B, TE, H]
                 const h16* __restrict__ Wihf,          // [GP][HP] f16 padded
                 const h16* __restrict__ Whhf,          // [GP][HP]
                 const float* __restrict__ bih, const float* __restrict__ bhh,
                 h16* __restrict__ hseq,                // [B, TE, H] f16 out
                 float* __restrict__ hfinal) {          // [B, H] f32 out
  extern __shared__ char smem[];
  h16*   xb = (h16*)smem;              // [16][HP]
  h16*   hb = xb + 16 * HP;            // [16][HP]
  float* hf = (float*)(hb + 16 * HP);  // [16][H]
  float* gi = hf + 16 * H;             // [16][GP]
  float* gh = gi + 16 * GP;            // [16][GP]
  int lane = threadIdx.x & 31;
  int b0 = blockIdx.x * 16;

  // warm L2/WGP$ for the weight images
  __builtin_prefetch(Wihf + lane * 64, 0, 3);
  __builtin_prefetch(Whhf + lane * 64, 0, 3);

  for (int i = lane; i < 16 * HP; i += 32) { xb[i] = (h16)0.f; hb[i] = (h16)0.f; }
  for (int i = lane; i < 16 * H;  i += 32) hf[i] = 0.f;
  __syncthreads();

  for (int t = 0; t < TE; ++t) {
    for (int e = lane; e < 16 * H; e += 32) {
      int b = e / H, c = e - b * H;
      xb[b * HP + c] = (h16)(float)xseq[(size_t)(b0 + b) * (TE * H) + t * H + c];
      if (t + 1 < TE)
        __builtin_prefetch(&xseq[(size_t)(b0 + b) * (TE * H) + (t + 1) * H + c], 0, 3);
    }
    __syncthreads();
    wave_gemm<4>(xb, HP, Wihf, gi, GP, GP / 16);
    wave_gemm<4>(hb, HP, Whhf, gh, GP, GP / 16);
    __syncthreads();
    gru_elem(gi, gh, bih, bhh, hf, hb,
             hseq + (size_t)b0 * (TE * H) + t * H, TE * H);
    __syncthreads();
  }
  for (int e = lane; e < 16 * H; e += 32) {
    int b = e / H, c = e - b * H;
    hfinal[(size_t)(b0 + b) * H + c] = hf[b * H + c];
  }
}

// ---------------- energy = enc @ attn_W^T + b  (rows = B*TE) ----------------
__global__ void __launch_bounds__(32)
energy_kernel(const h16* __restrict__ enc, const h16* __restrict__ Wf, // [112][HP]
              const float* __restrict__ bias, h16* __restrict__ energy, int nrows) {
  extern __shared__ char smem[];
  h16*   ab = (h16*)smem;               // [16][HP]
  float* cb = (float*)(ab + 16 * HP);   // [16][112]
  int lane = threadIdx.x & 31;
  int row0 = blockIdx.x * 16;
  if (row0 >= nrows) return;
  for (int i = lane; i < 16 * HP; i += 32) ab[i] = (h16)0.f;
  __syncthreads();
  for (int e = lane; e < 16 * H; e += 32) {
    int r = e / H, c = e - r * H;
    ab[r * HP + c] = enc[(size_t)(row0 + r) * H + c];
  }
  __syncthreads();
  wave_gemm<4>(ab, HP, Wf, cb, 112, 7);
  __syncthreads();
  for (int e = lane; e < 16 * H; e += 32) {
    int r = e / H, c = e - r * H;
    energy[(size_t)(row0 + r) * H + c] = (h16)(cb[r * 112 + c] + bias[c]);
  }
}

// ---------------- fused decoder: 50 steps attention + 2-layer GRU + out + MLP ---------
__global__ void __launch_bounds__(32)
decoder_kernel(const float* __restrict__ p,              // [B, TD, H]
               const h16* __restrict__ enc_g,            // [B*TE, H] f16
               const h16* __restrict__ eng_g,            // [B*TE, H] f16
               const float* __restrict__ h0i, const float* __restrict__ h1i,
               const h16* __restrict__ Wih0f,            // [GP][K2P]
               const h16* __restrict__ Whh0f,            // [GP][HP]
               const h16* __restrict__ Wih1f,            // [GP][HP]
               const h16* __restrict__ Whh1f,            // [GP][HP]
               const float* __restrict__ bih0, const float* __restrict__ bhh0,
               const float* __restrict__ bih1, const float* __restrict__ bhh1,
               const h16* __restrict__ outWf,            // [64][K2P]
               const float* __restrict__ outb,
               const float* __restrict__ f1W, const float* __restrict__ f1b,
               const float* __restrict__ f2W, const float* __restrict__ f2b,
               const float* __restrict__ f3W, const float* __restrict__ f3b,
               const float* __restrict__ f4W, const float* __restrict__ f4b,
               float* __restrict__ out) {                // [B, 2]
  extern __shared__ char smem[];
  float* enc_l = (float*)smem;            // [16][TE][H]
  float* eng_l = enc_l + 16 * TE * H;     // [16][TE][H]
  float* gi    = eng_l + 16 * TE * H;     // [16][GP]
  float* gh    = gi + 16 * GP;            // [16][GP]
  float* h0f   = gh + 16 * GP;            // [16][H]
  float* h1f   = h0f + 16 * H;            // [16][H]
  float* ctx   = h1f + 16 * H;            // [16][H]
  float* aw    = ctx + 16 * H;            // [16][TE]
  float* sbuf  = aw + 16 * TE;            // [16][64]
  h16*   ab    = (h16*)(sbuf + 16 * 64);  // [16][K2P]
  h16*   h0h   = ab + 16 * K2P;           // [16][HP]
  h16*   h1h   = h0h + 16 * HP;           // [16][HP]

  int lane = threadIdx.x & 31;
  int b0 = blockIdx.x * 16;

  // warm caches for loop-invariant weight images
  __builtin_prefetch(Wih0f + lane * 64, 0, 3);
  __builtin_prefetch(Whh0f + lane * 64, 0, 3);
  __builtin_prefetch(Wih1f + lane * 64, 0, 3);
  __builtin_prefetch(Whh1f + lane * 64, 0, 3);

  for (int i = lane; i < 16 * K2P; i += 32) ab[i] = (h16)0.f;
  for (int i = lane; i < 16 * HP;  i += 32) { h0h[i] = (h16)0.f; h1h[i] = (h16)0.f; }
  for (int e = lane; e < 16 * TE * H; e += 32) {
    int b = e / (TE * H), rem = e - b * (TE * H);
    enc_l[e] = (float)enc_g[(size_t)(b0 + b) * (TE * H) + rem];
    eng_l[e] = (float)eng_g[(size_t)(b0 + b) * (TE * H) + rem];
  }
  for (int e = lane; e < 16 * H; e += 32) {
    int b = e / H, c = e - b * H;
    float a0 = h0i[(size_t)(b0 + b) * H + c];
    float a1 = h1i[(size_t)(b0 + b) * H + c];
    h0f[e] = a0; h1f[e] = a1;
    h0h[b * HP + c] = (h16)a0;
    h1h[b * HP + c] = (h16)a1;
  }
  __syncthreads();

  for (int t = 0; t < TD; ++t) {
    // attention scores: scores[b][tt] = h1 . energy[b][tt]  (batched -> VALU)
    for (int e = lane; e < 16 * TE; e += 32) {
      int b = e / TE, tt = e - b * TE;
      const float* hp = h1f + b * H;
      const float* ep = eng_l + (b * TE + tt) * H;
      float s = 0.f;
      for (int d = 0; d < H; ++d) s += hp[d] * ep[d];
      aw[e] = s;
    }
    __syncthreads();
    if (lane < 16) {  // softmax over TE=12 per row
      float* r = aw + lane * TE;
      float m = r[0];
#pragma unroll
      for (int i = 1; i < TE; ++i) m = fmaxf(m, r[i]);
      float sum = 0.f;
#pragma unroll
      for (int i = 0; i < TE; ++i) { float v = __expf(r[i] - m); r[i] = v; sum += v; }
      float inv = 1.0f / sum;
#pragma unroll
      for (int i = 0; i < TE; ++i) r[i] *= inv;
    }
    __syncthreads();
    // context + build A-fragment source [p_t | ctx]
    for (int e = lane; e < 16 * H; e += 32) {
      int b = e / H, d = e - b * H;
      const float* awr = aw + b * TE;
      const float* er  = enc_l + b * TE * H + d;
      float s = 0.f;
#pragma unroll
      for (int tt = 0; tt < TE; ++tt) s += awr[tt] * er[tt * H];
      ctx[e] = s;
      ab[b * K2P + H + d] = (h16)s;
      ab[b * K2P + d]     = (h16)p[(size_t)(b0 + b) * (TD * H) + t * H + d];
      if (t + 1 < TD)
        __builtin_prefetch(&p[(size_t)(b0 + b) * (TD * H) + (t + 1) * H + d], 0, 3);
    }
    __syncthreads();
    // layer 0
    wave_gemm<7>(ab,  K2P, Wih0f, gi, GP, GP / 16);
    wave_gemm<4>(h0h, HP,  Whh0f, gh, GP, GP / 16);
    __syncthreads();
    gru_elem(gi, gh, bih0, bhh0, h0f, h0h, (h16*)nullptr, 0);
    __syncthreads();
    // layer 1
    wave_gemm<4>(h0h, HP, Wih1f, gi, GP, GP / 16);
    wave_gemm<4>(h1h, HP, Whh1f, gh, GP, GP / 16);
    __syncthreads();
    gru_elem(gi, gh, bih1, bhh1, h1f, h1h, (h16*)nullptr, 0);
    __syncthreads();
  }

  // out_context at final step: ab cols [100..199] still hold ctx(t=49); put h1 in [0..99]
  for (int e = lane; e < 16 * H; e += 32) {
    int b = e / H, d = e - b * H;
    ab[b * K2P + d] = (h16)h1f[e];
  }
  __syncthreads();
  wave_gemm<7>(ab, K2P, outWf, sbuf, 64, 4);
  __syncthreads();
  for (int e = lane; e < 16 * 50; e += 32) {
    int b = e / 50, c = e - b * 50;
    sbuf[b * 64 + c] = fmaxf(sbuf[b * 64 + c] + outb[c], 0.f);
  }
  __syncthreads();

  // tiny MLP 50->25->10->4->2 + softmax, one batch row per lane (lanes 0..15)
  if (lane < 16) {
    int b = lane;
    float v1[25];
#pragma unroll
    for (int i = 0; i < 25; ++i) {
      float s = f1b[i];
      for (int j = 0; j < 50; ++j) s += f1W[i * 50 + j] * sbuf[b * 64 + j];
      v1[i] = fmaxf(s, 0.f);
    }
    float v2[10];
#pragma unroll
    for (int i = 0; i < 10; ++i) {
      float s = f2b[i];
#pragma unroll
      for (int j = 0; j < 25; ++j) s += f2W[i * 25 + j] * v1[j];
      v2[i] = fmaxf(s, 0.f);
    }
    float v3[4];
#pragma unroll
    for (int i = 0; i < 4; ++i) {
      float s = f3b[i];
#pragma unroll
      for (int j = 0; j < 10; ++j) s += f3W[i * 10 + j] * v2[j];
      v3[i] = fmaxf(s, 0.f);
    }
    float v4[2];
#pragma unroll
    for (int i = 0; i < 2; ++i) {
      float s = f4b[i];
#pragma unroll
      for (int j = 0; j < 4; ++j) s += f4W[i * 4 + j] * v3[j];
      v4[i] = s;
    }
    float m = fmaxf(v4[0], v4[1]);
    float e0 = __expf(v4[0] - m), e1 = __expf(v4[1] - m);
    float inv = 1.f / (e0 + e1);
    out[(size_t)(b0 + b) * 2 + 0] = e0 * inv;
    out[(size_t)(b0 + b) * 2 + 1] = e1 * inv;
  }
}

// ---------------- host ----------------
extern "C" void kernel_launch(void* const* d_in, const int* in_sizes, int n_in,
                              void* d_out, int out_size, void* d_ws, size_t ws_size,
                              hipStream_t stream) {
  const float* q      = (const float*)d_in[0];
  const float* p      = (const float*)d_in[1];
  const float* attn_W = (const float*)d_in[2];
  const float* attn_b = (const float*)d_in[3];
  const float* eWih0  = (const float*)d_in[4];
  const float* eWhh0  = (const float*)d_in[5];
  const float* ebih0  = (const float*)d_in[6];
  const float* ebhh0  = (const float*)d_in[7];
  const float* eWih1  = (const float*)d_in[8];
  const float* eWhh1  = (const float*)d_in[9];
  const float* ebih1  = (const float*)d_in[10];
  const float* ebhh1  = (const float*)d_in[11];
  const float* dWih0  = (const float*)d_in[12];
  const float* dWhh0  = (const float*)d_in[13];
  const float* dbih0  = (const float*)d_in[14];
  const float* dbhh0  = (const float*)d_in[15];
  const float* dWih1  = (const float*)d_in[16];
  const float* dWhh1  = (const float*)d_in[17];
  const float* dbih1  = (const float*)d_in[18];
  const float* dbhh1  = (const float*)d_in[19];
  const float* outW   = (const float*)d_in[20];
  const float* outb   = (const float*)d_in[21];
  const float* f1W = (const float*)d_in[22]; const float* f1b = (const float*)d_in[23];
  const float* f2W = (const float*)d_in[24]; const float* f2b = (const float*)d_in[25];
  const float* f3W = (const float*)d_in[26]; const float* f3b = (const float*)d_in[27];
  const float* f4W = (const float*)d_in[28]; const float* f4b = (const float*)d_in[29];
  float* outp = (float*)d_out;

  const int B = in_sizes[0] / (TE * H);

  char* ws = (char*)d_ws;
  size_t off = 0;
  auto take = [&](size_t bytes) -> char* {
    char* r = ws + off;
    off = (off + bytes + 255) & ~(size_t)255;
    return r;
  };
  h16* attnWf = (h16*)take((size_t)112 * HP * 2);
  h16* eWih0f = (h16*)take((size_t)GP * HP * 2);
  h16* eWhh0f = (h16*)take((size_t)GP * HP * 2);
  h16* eWih1f = (h16*)take((size_t)GP * HP * 2);
  h16* eWhh1f = (h16*)take((size_t)GP * HP * 2);
  h16* dWih0f = (h16*)take((size_t)GP * K2P * 2);
  h16* dWhh0f = (h16*)take((size_t)GP * HP * 2);
  h16* dWih1f = (h16*)take((size_t)GP * HP * 2);
  h16* dWhh1f = (h16*)take((size_t)GP * HP * 2);
  h16* outWf  = (h16*)take((size_t)64 * K2P * 2);
  h16* h0seq  = (h16*)take((size_t)B * TE * H * 2);
  h16* encB   = (h16*)take((size_t)B * TE * H * 2);
  h16* energy = (h16*)take((size_t)B * TE * H * 2);
  float* h0i  = (float*)take((size_t)B * H * 4);
  float* h1i  = (float*)take((size_t)B * H * 4);
  (void)ws_size; (void)n_in; (void)out_size;

  auto pad = [&](const float* src, h16* dst, int N, int K, int NP, int KP) {
    int total = NP * KP;
    pad_f16<<<(total + 255) / 256, 256, 0, stream>>>(src, dst, N, K, NP, KP);
  };
  pad(attn_W, attnWf, H, H, 112, HP);
  pad(eWih0, eWih0f, G, H, GP, HP);
  pad(eWhh0, eWhh0f, G, H, GP, HP);
  pad(eWih1, eWih1f, G, H, GP, HP);
  pad(eWhh1, eWhh1f, G, H, GP, HP);
  pad(dWih0, dWih0f, G, K2, GP, K2P);
  pad(dWhh0, dWhh0f, G, H, GP, HP);
  pad(dWih1, dWih1f, G, H, GP, HP);
  pad(dWhh1, dWhh1f, G, H, GP, HP);
  pad(outW,  outWf, 50, K2, 64, K2P);

  const int tiles = B / 16;
  const size_t smem_gru = (size_t)16 * HP * 2 * 2 + (size_t)16 * H * 4 + (size_t)16 * GP * 4 * 2;
  const size_t smem_eng = (size_t)16 * HP * 2 + (size_t)16 * 112 * 4;
  const size_t smem_dec = (size_t)(16 * TE * H) * 4 * 2   // enc_l, eng_l
                        + (size_t)(16 * GP) * 4 * 2       // gi, gh
                        + (size_t)(16 * H) * 4 * 3        // h0f, h1f, ctx
                        + (size_t)(16 * TE) * 4           // aw
                        + (size_t)(16 * 64) * 4           // sbuf
                        + (size_t)(16 * K2P) * 2          // ab
                        + (size_t)(16 * HP) * 2 * 2;      // h0h, h1h

  hipFuncSetAttribute((const void*)decoder_kernel,
                      hipFuncAttributeMaxDynamicSharedMemorySize, (int)smem_dec);

  // Encoder layer 0 (input = q, fp32)
  gru_layer_kernel<float><<<tiles, 32, smem_gru, stream>>>(
      q, eWih0f, eWhh0f, ebih0, ebhh0, h0seq, h0i);
  // Encoder layer 1 (input = h0seq f16) -> enc
  gru_layer_kernel<h16><<<tiles, 32, smem_gru, stream>>>(
      h0seq, eWih1f, eWhh1f, ebih1, ebhh1, encB, h1i);
  // Attention energies
  energy_kernel<<<(B * TE) / 16, 32, smem_eng, stream>>>(encB, attnWf, attn_b, energy, B * TE);
  // Fused decoder + MLP
  decoder_kernel<<<tiles, 32, smem_dec, stream>>>(
      p, encB, energy, h0i, h1i,
      dWih0f, dWhh0f, dWih1f, dWhh1f,
      dbih0, dbhh0, dbih1, dbhh1,
      outWf, outb,
      f1W, f1b, f2W, f2b, f3W, f3b, f4W, f4b,
      outp);
}
